// GraphEncoder_74852690035138
// MI455X (gfx1250) — matrix operand order repaired and checked by reading, not compiled, not run on hardware
//
#include <hip/hip_runtime.h>
#include <hip/hip_bf16.h>

// ---------------------------------------------------------------------------
// GraphEncoder for MI455X (gfx1250, wave32, WMMA + async LDS DMA)
// All dense math goes through v_wmma_f32_16x16x32_f16 (f16 in, f32 acc).
// GEMM B-operand staged in LDS via GLOBAL_LOAD_ASYNC_TO_LDS_B128 (ASYNCcnt),
// double-buffered so DMA of stage i+1 overlaps WMMA of stage i.
// ---------------------------------------------------------------------------

typedef __attribute__((ext_vector_type(16))) _Float16 v16h;
typedef __attribute__((ext_vector_type(8)))  float    v8f;

#define HIDN   512
#define HHALF  256
#define BATCH  64
#define SEQL   256
#define NNODES 16384
#define NSAMP  10
#define EMBD   300
#define EMBP   320

#define BPITCH 40   // LDS B-row pitch in halves (80B): 20-dword stride, no bank conflicts

union U16 { v16h v; uint4 q[2]; };

// A-matrix 16x32 f16 tile (rows m0..m0+15, cols k0..k0+31), row-major src.
// ISA layout: lanes 0-15 -> M=lane, elements 0..7 = K k0..k0+7, 8..15 = K k0+16..k0+23
//             lanes 16-31 -> M=lane-16, K k0+8..k0+15 and k0+24..k0+31
__device__ inline v16h load_tile_a(const _Float16* __restrict__ base, int ld,
                                   int m0, int k0, int lane) {
    int half = lane >> 4;
    int row  = m0 + (lane & 15);
    int ka   = k0 + half * 8;
    U16 u;
    u.q[0] = *(const uint4*)(base + (size_t)row * ld + ka);
    u.q[1] = *(const uint4*)(base + (size_t)row * ld + ka + 16);
    return u.v;
}

// B-matrix 32x16 f16 tile for D = A x B with B[k][n] = W[n0+n][k0+k] (W row-major [N,K]).
// ISA layout: lane<16 -> N=lane, K k0..k0+7, k0+16..k0+23; lane>=16 -> N=lane-16, K k0+8..k0+15, k0+24..k0+31
__device__ inline v16h load_tile_b(const _Float16* __restrict__ base, int ld,
                                   int n0, int k0, int lane) {
    int row = n0 + (lane & 15);
    int ks  = k0 + (lane >> 4) * 8;
    U16 u;
    u.q[0] = *(const uint4*)(base + (size_t)row * ld + ks);
    u.q[1] = *(const uint4*)(base + (size_t)row * ld + ks + 16);
    return u.v;
}

__device__ inline float sigmf(float x) { return 1.f / (1.f + __expf(-x)); }

// Async 16B copy global -> LDS (tracked by ASYNCcnt, not LOADcnt).
__device__ inline void async_b128_to_lds(unsigned lds_off, const void* gptr) {
    asm volatile("global_load_async_to_lds_b128 %0, %1, off"
                 :: "v"(lds_off), "v"(gptr) : "memory");
}

// ---------------------------------------------------------------------------
// Generic GEMM: D[M,N](f32) = A[M,K](f16) * W[N,K]^T(f16) + bias[N], opt relu.
// Block = 256 threads = 8 waves. Block tile 128(M) x 64(N).
// B (64 rows x 32 K-halves = 4KB) is DMA'd to LDS once per k-step by the whole
// block (one b128 chunk per thread) instead of 8x redundantly per wave;
// double-buffered with s_wait_asynccnt for DMA/WMMA overlap.
// K multiple of 32, M multiple of 128, N multiple of 64.
// ---------------------------------------------------------------------------
__global__ __launch_bounds__(256) void gemm_f16(
    const _Float16* __restrict__ A, const _Float16* __restrict__ W,
    const float* __restrict__ bias, float* __restrict__ D,
    int M, int N, int K, int relu)
{
    __shared__ _Float16 bsh[2][64 * BPITCH];   // 2 x 5 KB staging buffers

    int lane = threadIdx.x & 31;
    int wave = threadIdx.x >> 5;
    int m0 = blockIdx.x * 128 + wave * 16;
    int n0 = blockIdx.y * 64;

    // per-thread DMA assignment: thread -> (row 0..63, 16B chunk 0..3)
    int crow = threadIdx.x >> 2;
    int cchk = threadIdx.x & 3;
    const _Float16* gsrc = W + (size_t)(n0 + crow) * K + cchk * 8;
    // LDS byte offset of a generic pointer is its low 32 bits (aperture rule)
    unsigned ldsB0  = (unsigned)(uintptr_t)(&bsh[0][0]);
    unsigned ldsRow = (unsigned)(crow * BPITCH + cchk * 8) * 2u;
    const unsigned bufBytes = 64 * BPITCH * 2;

    v8f acc[4];
    for (int g = 0; g < 4; ++g)
        for (int e = 0; e < 8; ++e) acc[g][e] = 0.f;

    const int nst = K / 32;
    async_b128_to_lds(ldsB0 + ldsRow, gsrc);              // stage 0 -> buf 0

    for (int i = 0; i < nst; ++i) {
        int k0 = i * 32;
        if (i + 1 < nst) {
            // issue next stage into the other buffer (last read at stage i-1,
            // protected by the end-of-stage barrier)
            async_b128_to_lds(ldsB0 + ((i + 1) & 1) * bufBytes + ldsRow,
                              gsrc + k0 + 32);
            asm volatile("s_wait_asynccnt 0x1" ::: "memory");
        } else {
            asm volatile("s_wait_asynccnt 0x0" ::: "memory");
        }
        if (k0 + 32 < K)  // warm next A cacheline (global_prefetch_b8)
            __builtin_prefetch(A + (size_t)(m0 + (lane & 15)) * K + k0 + 32, 0, 1);
        __syncthreads();   // all waves' DMA for buf[i&1] complete

        v16h a = load_tile_a(A, K, m0, k0, lane);
        const _Float16* bb = &bsh[i & 1][0];
        for (int g = 0; g < 4; ++g) {
            v16h b = load_tile_b(bb, BPITCH, g * 16, 0, lane);
            acc[g] = __builtin_amdgcn_wmma_f32_16x16x32_f16(
                false, a, false, b, (short)0, acc[g], false, false);
        }
        __syncthreads();   // reads of buf[i&1] done before it is overwritten
    }

    int me = (lane >> 4) * 8;      // D layout: VGPR e -> M = m0+e (+8 for hi lanes)
    int nl = lane & 15;
    for (int g = 0; g < 4; ++g) {
        int n = n0 + g * 16 + nl;
        float bv = bias ? bias[n] : 0.f;
        for (int e = 0; e < 8; ++e) {
            float v = acc[g][e] + bv;
            if (relu) v = fmaxf(v, 0.f);
            D[(size_t)(m0 + e + me) * N + n] = v;
        }
    }
}

// ---------------------------------------------------------------------------
// LSTM recurrent scan: one workgroup (32 waves) per direction.
// h kept in LDS as f16 (WMMA A operand); c kept in registers in the WMMA
// accumulator layout. Per step: G = h @ W_hh^T via WMMA, gates, update.
// W_hh (512KB f16) is L2-resident and streamed each step.
// gridDim.x = 2: block 0 = forward, block 1 = backward (reverse time).
// ---------------------------------------------------------------------------
__global__ __launch_bounds__(1024) void lstm_scan(
    const float* __restrict__ xw_f, const float* __restrict__ xw_b,
    const _Float16* __restrict__ whh_f, const _Float16* __restrict__ whh_b,
    float* outF32, _Float16* outF16)
{
    __shared__ _Float16 hsh[BATCH * HHALF];   // 64x256 f16 = 32 KB

    const int reverse = blockIdx.x;
    const float*    xw  = reverse ? xw_b  : xw_f;
    const _Float16* whh = reverse ? whh_b : whh_f;
    const int colOff    = reverse ? HHALF : 0;

    int lane = threadIdx.x & 31;
    int wave = threadIdx.x >> 5;
    int jg = wave >> 1;            // 16 hidden-column groups of 16
    int mh = wave & 1;             // batch halves: rows 0..31 / 32..63
    int j0 = jg * 16;
    int me = (lane >> 4) * 8;
    int jn = j0 + (lane & 15);

    for (int i = threadIdx.x; i < BATCH * HHALF; i += 1024) hsh[i] = (_Float16)0.f;
    __syncthreads();

    float c[2][8];
    for (int it = 0; it < 2; ++it)
        for (int e = 0; e < 8; ++e) c[it][e] = 0.f;

    for (int s = 0; s < SEQL; ++s) {
        int t = reverse ? (SEQL - 1 - s) : s;

        // init accumulators with x@W_ih^T + b (precomputed)
        v8f acc[2][4];
        for (int it = 0; it < 2; ++it) {
            int m0 = mh * 32 + it * 16;
            for (int g = 0; g < 4; ++g) {
                int n = g * HHALF + jn;
                for (int e = 0; e < 8; ++e)
                    acc[it][g][e] = xw[((size_t)(m0 + e + me) * SEQL + t) * 1024 + n];
            }
        }
        // G += h @ W_hh^T
        for (int kk = 0; kk < HHALF; kk += 32) {
            v16h bt[4];
            for (int g = 0; g < 4; ++g)
                bt[g] = load_tile_b(whh, HHALF, g * HHALF + j0, kk, lane);
            for (int it = 0; it < 2; ++it) {
                int m0 = mh * 32 + it * 16;
                v16h a = load_tile_a(hsh, HHALF, m0, kk, lane);
                for (int g = 0; g < 4; ++g)
                    acc[it][g] = __builtin_amdgcn_wmma_f32_16x16x32_f16(
                        false, a, false, bt[g], (short)0, acc[it][g], false, false);
            }
        }
        __syncthreads();   // all reads of h done before overwrite

        for (int it = 0; it < 2; ++it) {
            int m0 = mh * 32 + it * 16;
            for (int e = 0; e < 8; ++e) {
                int m = m0 + e + me;
                float iv = sigmf(acc[it][0][e]);
                float fv = sigmf(acc[it][1][e]);
                float gv = tanhf(acc[it][2][e]);
                float ov = sigmf(acc[it][3][e]);
                float cc = fv * c[it][e] + iv * gv;
                c[it][e] = cc;
                float h = ov * tanhf(cc);
                hsh[m * HHALF + jn] = (_Float16)h;
                size_t oi = ((size_t)m * SEQL + t) * HIDN + colOff + jn;
                if (outF32) outF32[oi] = h;
                if (outF16) outF16[oi] = (_Float16)h;
            }
        }
        __syncthreads();   // new h visible to all waves
    }
}

// ---------------------------------------------------------------------------
// Small helper kernels
// ---------------------------------------------------------------------------
__global__ void k_convert_pad(const float* __restrict__ src, _Float16* __restrict__ dst,
                              int rows, int K, int Kp) {
    int i = blockIdx.x * blockDim.x + threadIdx.x;
    if (i >= rows * Kp) return;
    int r = i / Kp, k = i % Kp;
    dst[i] = (_Float16)((k < K) ? src[(size_t)r * K + k] : 0.f);
}

__global__ void k_bias_sum(const float* a, const float* b, float* o, int n) {
    int i = blockIdx.x * blockDim.x + threadIdx.x;
    if (i < n) o[i] = a[i] + b[i];
}

__global__ void k_embed(const int* __restrict__ feat, const float* __restrict__ emb,
                        _Float16* __restrict__ x) {
    int i = blockIdx.x * blockDim.x + threadIdx.x;
    if (i >= NNODES * EMBP) return;
    int row = i / EMBP, k = i % EMBP;
    int tok = feat[row];   // feature_info[:-1].reshape -> first 64*256 entries
    x[i] = (_Float16)((k < EMBD) ? emb[(size_t)tok * EMBD + k] : 0.f);
}

__global__ void k_maxtime(const float* __restrict__ in, float* __restrict__ out, int C) {
    int i = blockIdx.x * blockDim.x + threadIdx.x;
    if (i >= BATCH * C) return;
    int b = i / C, ch = i % C;
    float m = in[((size_t)b * SEQL) * C + ch];
    for (int t = 1; t < SEQL; ++t)
        m = fmaxf(m, in[((size_t)b * SEQL + t) * C + ch]);
    out[i] = m;
}

__global__ void k_repres(const float* __restrict__ ov, const float* __restrict__ pad,
                         float* __restrict__ repres) {
    int i = blockIdx.x * blockDim.x + threadIdx.x;
    if (i >= (NNODES + 1) * HIDN) return;
    repres[i] = (i < NNODES * HIDN) ? ov[i] : pad[i - NNODES * HIDN];
}

// one wave per node: len = #neighbors with any positive component
__global__ void k_len(const int* __restrict__ adj, const int* __restrict__ nodes,
                      const float* __restrict__ repres, float* __restrict__ len) {
    int gw   = (blockIdx.x * blockDim.x + threadIdx.x) >> 5;
    int lane = threadIdx.x & 31;
    if (gw >= NNODES) return;
    int node = nodes[gw];
    int cnt = 0;
    for (int s = 0; s < NSAMP; ++s) {
        int nb = adj[(size_t)node * NSAMP + s];
        int pos = 0;
        for (int d = lane; d < HIDN; d += 32)
            pos |= (repres[(size_t)nb * HIDN + d] > 0.f);
        if (__ballot(pos) != 0ull) cnt++;
    }
    if (lane == 0) len[gw] = (float)cnt;
}

__global__ void k_init_hidden(const int* __restrict__ nodes, const float* __restrict__ repres,
                              float* __restrict__ fwh, float* __restrict__ bwh) {
    int i = blockIdx.x * blockDim.x + threadIdx.x;
    if (i >= NNODES * HIDN) return;
    int r = i / HIDN, d = i % HIDN;
    float v = repres[(size_t)nodes[r] * HIDN + d];
    fwh[i] = v; bwh[i] = v;
}

// comb[i] = [ self(512) | masked-mean of neighbors(512) ] as f16
__global__ void k_comb(const float* __restrict__ selfH, const float* __restrict__ src,
                       const float* __restrict__ pad, const int* __restrict__ adj,
                       const int* __restrict__ nodes, const float* __restrict__ len,
                       _Float16* __restrict__ comb) {
    int i = blockIdx.x;       // node
    int d = threadIdx.x;      // 512 dims
    int node = nodes[i];
    float sum = 0.f;
    for (int s = 0; s < NSAMP; ++s) {
        int nb = adj[(size_t)node * NSAMP + s];
        float v = (nb < NNODES) ? src[(size_t)nb * HIDN + d]
                                : (pad ? pad[d] : 0.f);
        sum += v;
    }
    float mean = sum / fmaxf(len[i], 1.f);
    comb[(size_t)i * 1024 + d]        = (_Float16)selfH[(size_t)i * HIDN + d];
    comb[(size_t)i * 1024 + HIDN + d] = (_Float16)mean;
}

__global__ void k_assemble(const float* __restrict__ fwh, const float* __restrict__ bwh,
                           float* __restrict__ out) {
    int i = blockIdx.x * blockDim.x + threadIdx.x;
    if (i >= NNODES * HIDN) return;
    int r = i / HIDN, d = i % HIDN;
    out[(size_t)r * 1024 + d]        = fwh[i];
    out[(size_t)r * 1024 + 512 + d]  = bwh[i];
}

// ---------------------------------------------------------------------------
extern "C" void kernel_launch(void* const* d_in, const int* in_sizes, int n_in,
                              void* d_out, int out_size, void* d_ws, size_t ws_size,
                              hipStream_t stream) {
    (void)in_sizes; (void)n_in; (void)out_size; (void)ws_size;

    const int*   fw_adj = (const int*)d_in[0];
    const int*   bw_adj = (const int*)d_in[1];
    const int*   feat   = (const int*)d_in[2];
    const int*   nodes  = (const int*)d_in[3];
    const float* emb    = (const float*)d_in[4];
    // lstm_params flattened: [layer][dir] of (W_ih, W_hh, b_ih, b_hh)
    const float* Wih[2][2] = {{(const float*)d_in[5],  (const float*)d_in[9]},
                              {(const float*)d_in[13], (const float*)d_in[17]}};
    const float* Whh[2][2] = {{(const float*)d_in[6],  (const float*)d_in[10]},
                              {(const float*)d_in[14], (const float*)d_in[18]}};
    const float* bih[2][2] = {{(const float*)d_in[7],  (const float*)d_in[11]},
                              {(const float*)d_in[15], (const float*)d_in[19]}};
    const float* bhh[2][2] = {{(const float*)d_in[8],  (const float*)d_in[12]},
                              {(const float*)d_in[16], (const float*)d_in[20]}};
    const float* padvec = (const float*)d_in[21];
    const float* fwW = (const float*)d_in[22];
    const float* fwb = (const float*)d_in[23];
    const float* bwW = (const float*)d_in[24];
    const float* bwb = (const float*)d_in[25];

    float* out        = (float*)d_out;
    float* out_ge     = out;                              // [64,1024] == [128,512]
    float* out_hidden = out + 65536;                      // [64,256,1024]
    float* out_seq    = out + 65536 + 16777216;           // [64,512]
    float* out_ov     = out + 65536 + 16777216 + 32768;   // [64,256,512]

    // ---- workspace layout -------------------------------------------------
    char* ws = (char*)d_ws; size_t off = 0;
    auto alloc = [&](size_t b) -> char* {
        char* p = ws + off; off = (off + b + 255) & ~(size_t)255; return p;
    };
    _Float16* wih16[2][2];
    wih16[0][0] = (_Float16*)alloc((size_t)1024 * EMBP * 2);
    wih16[0][1] = (_Float16*)alloc((size_t)1024 * EMBP * 2);
    wih16[1][0] = (_Float16*)alloc((size_t)1024 * 512 * 2);
    wih16[1][1] = (_Float16*)alloc((size_t)1024 * 512 * 2);
    _Float16* whh16[2][2];
    for (int l = 0; l < 2; ++l)
        for (int dd = 0; dd < 2; ++dd)
            whh16[l][dd] = (_Float16*)alloc((size_t)1024 * HHALF * 2);
    float* bsum[2][2];
    for (int l = 0; l < 2; ++l)
        for (int dd = 0; dd < 2; ++dd)
            bsum[l][dd] = (float*)alloc(1024 * 4);
    _Float16* sageW16[2];
    sageW16[0] = (_Float16*)alloc((size_t)3 * 512 * 1024 * 2);
    sageW16[1] = (_Float16*)alloc((size_t)3 * 512 * 1024 * 2);
    _Float16* x16     = (_Float16*)alloc((size_t)NNODES * EMBP * 2);
    _Float16* l0out16 = (_Float16*)alloc((size_t)NNODES * HIDN * 2);
    float* lenv[2];
    lenv[0] = (float*)alloc(NNODES * 4);
    lenv[1] = (float*)alloc(NNODES * 4);
    // big region: first xw[2] (2x64MB), later reused for GraphSAGE buffers
    char* big = alloc((size_t)132 * 1024 * 1024);
    float* xwb[2] = {(float*)big, (float*)(big + (size_t)64 * 1024 * 1024)};
    float*    repres = (float*)big;                                   // 32.0 MB
    float*    hid[2] = {(float*)(big + (size_t)34 * 1024 * 1024),
                        (float*)(big + (size_t)66 * 1024 * 1024)};    // 32 MB each
    _Float16* comb   = (_Float16*)(big + (size_t)98 * 1024 * 1024);   // 32 MB

    auto cdiv = [](long a, long b) { return (unsigned)((a + b - 1) / b); };

    // ---- 1. weight / bias conversion --------------------------------------
    for (int l = 0; l < 2; ++l)
        for (int dd = 0; dd < 2; ++dd) {
            int Kin = (l == 0) ? EMBD : 512;
            int Kp  = (l == 0) ? EMBP : 512;
            k_convert_pad<<<cdiv((long)1024 * Kp, 256), 256, 0, stream>>>(
                Wih[l][dd], wih16[l][dd], 1024, Kin, Kp);
            k_convert_pad<<<cdiv((long)1024 * HHALF, 256), 256, 0, stream>>>(
                Whh[l][dd], whh16[l][dd], 1024, HHALF, HHALF);
            k_bias_sum<<<4, 256, 0, stream>>>(bih[l][dd], bhh[l][dd], bsum[l][dd], 1024);
        }
    k_convert_pad<<<cdiv((long)1536 * 1024, 256), 256, 0, stream>>>(fwW, sageW16[0], 1536, 1024, 1024);
    k_convert_pad<<<cdiv((long)1536 * 1024, 256), 256, 0, stream>>>(bwW, sageW16[1], 1536, 1024, 1024);

    // ---- 2. embedding gather (f16, K padded 300->320) ----------------------
    k_embed<<<cdiv((long)NNODES * EMBP, 256), 256, 0, stream>>>(feat, emb, x16);

    // ---- 3. BiLSTM layer 0 --------------------------------------------------
    dim3 gx(128, 16);
    gemm_f16<<<gx, 256, 0, stream>>>(x16, wih16[0][0], bsum[0][0], xwb[0], NNODES, 1024, EMBP, 0);
    gemm_f16<<<gx, 256, 0, stream>>>(x16, wih16[0][1], bsum[0][1], xwb[1], NNODES, 1024, EMBP, 0);
    lstm_scan<<<2, 1024, 0, stream>>>(xwb[0], xwb[1], whh16[0][0], whh16[0][1],
                                      nullptr, l0out16);

    // ---- 4. BiLSTM layer 1 (output_vector -> d_out directly) ---------------
    gemm_f16<<<gx, 256, 0, stream>>>(l0out16, wih16[1][0], bsum[1][0], xwb[0], NNODES, 1024, 512, 0);
    gemm_f16<<<gx, 256, 0, stream>>>(l0out16, wih16[1][1], bsum[1][1], xwb[1], NNODES, 1024, 512, 0);
    lstm_scan<<<2, 1024, 0, stream>>>(xwb[0], xwb[1], whh16[1][0], whh16[1][1],
                                      out_ov, nullptr);

    // ---- 5. seq_embedding + node_repres ------------------------------------
    k_maxtime<<<cdiv((long)BATCH * 512, 256), 256, 0, stream>>>(out_ov, out_seq, 512);
    k_repres<<<cdiv((long)(NNODES + 1) * HIDN, 256), 256, 0, stream>>>(out_ov, padvec, repres);

    // ---- 6. GraphSAGE --------------------------------------------------------
    k_len<<<2048, 256, 0, stream>>>(fw_adj, nodes, repres, lenv[0]);
    k_len<<<2048, 256, 0, stream>>>(bw_adj, nodes, repres, lenv[1]);
    k_init_hidden<<<cdiv((long)NNODES * HIDN, 256), 256, 0, stream>>>(nodes, repres, hid[0], hid[1]);

    dim3 gs(128, 8);
    for (int l = 0; l < 3; ++l) {
        for (int dd = 0; dd < 2; ++dd) {
            const int*   adj = dd ? bw_adj : fw_adj;
            const float* src = (l == 0) ? repres : hid[dd];
            const float* pad = (l == 0) ? padvec : nullptr;
            const float* bp  = (dd ? bwb : fwb) + (size_t)l * 512;
            k_comb<<<NNODES, 512, 0, stream>>>(hid[dd], src, pad, adj, nodes, lenv[dd], comb);
            gemm_f16<<<gs, 256, 0, stream>>>(comb, sageW16[dd] + (size_t)l * 512 * 1024,
                                             bp, hid[dd], NNODES, 512, 1024, 1);
        }
    }

    // ---- 7. final outputs ----------------------------------------------------
    k_assemble<<<cdiv((long)NNODES * HIDN, 256), 256, 0, stream>>>(hid[0], hid[1], out_hidden);
    k_maxtime<<<cdiv((long)BATCH * 1024, 256), 256, 0, stream>>>(out_hidden, out_ge, 1024);
}
